// CrossCryptocurrencyAttention_11312943858195
// MI455X (gfx1250) — matrix operand order repaired and checked by reading, not compiled, not run on hardware
//
#include <hip/hip_runtime.h>
#include <hip/hip_bf16.h>

// ---------------------------------------------------------------------------
// Cross-crypto attention, fused for MI455X (gfx1250, wave32, WMMA).
// Phase 1: QKV projections (f32 -> f16, bias folded, Q pre-scaled, V stored
//          transposed [dim][seq] for wide P*V fragment loads)
// Phase 2: flash-style cross attention with DOUBLE-BUFFERED ASYNC K/V staging
//          (global_load_async_to_lds_b128 + s_wait_asynccnt) overlapping the
//          copy engine with WMMA math.
// Phase 3: output projection (f32 out)
// All matrix math via v_wmma_f32_16x16x32_f16; all LDS fragment traffic is
// ds_load_b128 / ds_store_b128 (layouts matched to ISA 7.12.2 fragments).
// ---------------------------------------------------------------------------

#define DM    256
#define SEQ   2048
#define BATCH 2
#define NH    8
#define HD    32

typedef __attribute__((ext_vector_type(16))) _Float16 v16h;
typedef __attribute__((ext_vector_type(8)))  _Float16 v8h;
typedef __attribute__((ext_vector_type(8)))  float    v8f;

#define CONCAT16(lo, hi) __builtin_shufflevector(lo, hi, 0,1,2,3,4,5,6,7,8,9,10,11,12,13,14,15)
#define ILV16(lo, hi)    __builtin_shufflevector(lo, hi, 0,8,1,9,2,10,3,11,4,12,5,13,6,14,7,15)

__device__ __forceinline__ v8h cvt_v8h(float4 a, float4 b) {
  v8h r;
  r[0] = (_Float16)a.x; r[1] = (_Float16)a.y; r[2] = (_Float16)a.z; r[3] = (_Float16)a.w;
  r[4] = (_Float16)b.x; r[5] = (_Float16)b.y; r[6] = (_Float16)b.z; r[7] = (_Float16)b.w;
  return r;
}

// low 32 bits of a generic pointer to LDS == LDS byte offset
__device__ __forceinline__ unsigned lds_addr(const void* p) {
  return (unsigned)(unsigned long long)(uintptr_t)p;
}

// async 16B copy global -> LDS (per-lane), tracked by ASYNCcnt
__device__ __forceinline__ void async_copy_b128(unsigned lds, const void* g) {
  asm volatile("global_load_async_to_lds_b128 %0, %1, off"
               :: "v"(lds), "v"((unsigned long long)(uintptr_t)g) : "memory");
}

// ---------------------------------------------------------------------------
// 128x64 tile GEMM core: C += A(128xK, f32->f16) * B(Kx64, f32->f16).
// 128 threads = 4 waves; wave w owns rows [w*32, w*32+32) as two 16-row
// A-fragments across 4 N-subtiles (8 WMMAs / wave / K-chunk).
// B tile kept transposed in LDS so B-fragments are contiguous b128 loads.
// ---------------------------------------------------------------------------
__device__ __forceinline__ void gemm128x64(const float* __restrict__ A,
                                           const float* __restrict__ B,
                                           int m0, int n0, int K, int lda, int ldb,
                                           int t, v8f c[2][4],
                                           _Float16 As[128][40], _Float16 BsT[64][40])
{
  const int wave = t >> 5, lane = t & 31;
  const int lm  = lane & 15;
  const int kb  = (lane >> 4) * 8;    // A-frag K base (0 or 8)
  const int kb2 = (lane >> 4) * 16;   // unused; B-frag also uses kb (0 or 8)

  for (int k0 = 0; k0 < K; k0 += 32) {
    // A tile 128x32: 8 contiguous floats -> v8h -> one ds_store_b128
    #pragma unroll
    for (int u = 0; u < 4; u++) {
      int uid = u * 128 + t;
      int r = uid >> 2, cg = (uid & 3) * 8;
      const float* src = A + (size_t)(m0 + r) * lda + k0 + cg;
      *(v8h*)&As[r][cg] = cvt_v8h(*(const float4*)src, *(const float4*)(src + 4));
    }
    // B tile 32x64, transposed into LDS: 8 coalesced dword loads -> one b128 store
    #pragma unroll
    for (int u = 0; u < 2; u++) {
      int uid = u * 128 + t;
      int col = uid >> 2, kg = (uid & 3) * 8;
      v8h pk;
      #pragma unroll
      for (int e = 0; e < 8; e++)
        pk[e] = (_Float16)B[(size_t)(k0 + kg + e) * ldb + n0 + col];
      *(v8h*)&BsT[col][kg] = pk;
    }
    __syncthreads();

    v16h af[2];
    #pragma unroll
    for (int s = 0; s < 2; s++) {
      const _Float16* ar = &As[wave * 32 + s * 16 + lm][0];
      v8h lo = *(const v8h*)(ar + kb);
      v8h hi = *(const v8h*)(ar + 16 + kb);
      af[s] = CONCAT16(lo, hi);
    }
    #pragma unroll
    for (int ns = 0; ns < 4; ns++) {
      const _Float16* br = &BsT[ns * 16 + lm][0];
      v8h lo = *(const v8h*)(br + kb);
      v8h hi = *(const v8h*)(br + 16 + kb);
      v16h bf = CONCAT16(lo, hi);
      c[0][ns] = __builtin_amdgcn_wmma_f32_16x16x32_f16(false, af[0], false, bf,
                                                        (short)0, c[0][ns], false, false);
      c[1][ns] = __builtin_amdgcn_wmma_f32_16x16x32_f16(false, af[1], false, bf,
                                                        (short)0, c[1][ns], false, false);
    }
    __syncthreads();
  }
}

// ---------------------------------------------------------------------------
// Phase 1: QKV projections. z = asset*3 + {0:q,1:k,2:v}. M=B*S=4096, N=K=256.
// Q/K written [asset][B][H][S][HD] f16 (Q pre-scaled by 1/sqrt(HD));
// V written transposed [asset][B][H][HD][S] f16.
// ---------------------------------------------------------------------------
__global__ __launch_bounds__(128) void qkv_proj_kernel(
    const float* __restrict__ x0, const float* __restrict__ x1, const float* __restrict__ x2,
    const float* __restrict__ Wq, const float* __restrict__ bq,
    const float* __restrict__ Wk, const float* __restrict__ bk,
    const float* __restrict__ Wv, const float* __restrict__ bv,
    _Float16* __restrict__ qws, _Float16* __restrict__ kws, _Float16* __restrict__ vws)
{
  __shared__ __align__(16) _Float16 As[128][40];
  __shared__ __align__(16) _Float16 BsT[64][40];

  const int z = blockIdx.z;
  const int a = z / 3, p = z % 3;
  const float* X    = (a == 0) ? x0 : (a == 1) ? x1 : x2;
  const float* W    = ((p == 0) ? Wq : (p == 1) ? Wk : Wv) + (size_t)a * DM * DM;
  const float* bias = ((p == 0) ? bq : (p == 1) ? bk : bv) + (size_t)a * DM;
  _Float16* dst = (p == 0) ? qws : (p == 1) ? kws : vws;

  const int m0 = blockIdx.x * 128;
  const int n0 = blockIdx.y * 64;
  const int t = threadIdx.x;
  const int wave = t >> 5, lane = t & 31;
  const int lm = lane & 15, rbase = (lane >> 4) * 8;

  v8f c[2][4] = {};
  gemm128x64(X, W, m0, n0, DM, DM, DM, t, c, As, BsT);

  const float qscale = 0.17677669529663687f;  // 1/sqrt(32), folded into Q
  #pragma unroll
  for (int s = 0; s < 2; s++) {
    #pragma unroll
    for (int ns = 0; ns < 4; ns++) {
      #pragma unroll
      for (int r = 0; r < 8; r++) {
        int row = m0 + wave * 32 + s * 16 + rbase + r;  // row = b*SEQ + seq
        int col = n0 + ns * 16 + lm;                    // col = h*HD + d
        float v = c[s][ns][r] + bias[col];
        if (p == 0) v *= qscale;
        int b  = row >> 11, sq = row & (SEQ - 1);
        int h  = col >> 5,  d  = col & (HD - 1);
        size_t base = (size_t)((a * BATCH + b) * NH + h) * SEQ * HD;
        if (p == 2)
          dst[base + (size_t)d * SEQ + sq] = (_Float16)v;   // V^T: [dim][seq]
        else
          dst[base + (size_t)sq * HD + d] = (_Float16)v;    // Q/K: [seq][dim]
      }
    }
  }
}

// ---------------------------------------------------------------------------
// Phase 2: fused cross attention. Block: 128 thr = 4 waves, 64 query rows.
// grid = (SEQ/64, B*H, 3 assets).  out_i = sum_j softmax_t(Q_i K_j^T) V_j.
// K/V chunks stream through a 2-deep LDS pipeline filled by async-LDS DMA;
// P is packed (key n, key n+16) per dword; the matching key permutation on
// the V side is a register interleave of two contiguous b128 LDS loads.
// ---------------------------------------------------------------------------
__global__ __launch_bounds__(128) void cross_attn_kernel(
    const _Float16* __restrict__ qws, const _Float16* __restrict__ kws,
    const _Float16* __restrict__ vws, float* __restrict__ ows)
{
  __shared__ __align__(16) _Float16 Ktb[2][32][40];   // [buf][key][dim]
  __shared__ __align__(16) _Float16 VtTb[2][32][40];  // [buf][dim][key]
  __shared__ __align__(16) unsigned Pt[4][16][20];    // per-wave 16 rows x 16 packed pairs

  const int i  = blockIdx.z;
  const int b  = blockIdx.y >> 3, h = blockIdx.y & 7;
  const int q0 = blockIdx.x * 64;
  const int t  = threadIdx.x;
  const int wave = t >> 5, lane = t & 31;
  const int lm  = lane & 15;
  const int rbase = (lane >> 4) * 8;
  const int kb  = (lane >> 4) * 8;
  const int kb2 = (lane >> 4) * 16;

  // this thread's slice of each 32x32 chunk (16B of K, 16B of V^T)
  const int crow = t >> 2, csub = (t & 3) * 8;
  const unsigned ldsK[2] = { lds_addr(&Ktb[0][crow][csub]),  lds_addr(&Ktb[1][crow][csub]) };
  const unsigned ldsV[2] = { lds_addr(&VtTb[0][crow][csub]), lds_addr(&VtTb[1][crow][csub]) };

  // Q A-fragment for this wave's 16 rows (scale folded in): 2 x 16B global loads
  const _Float16* Qb = qws + (size_t)((i * BATCH + b) * NH + h) * SEQ * HD;
  v16h qf;
  {
    const _Float16* qrow = Qb + (size_t)(q0 + wave * 16 + lm) * HD;
    v8h lo = *(const v8h*)(qrow + kb);
    v8h hi = *(const v8h*)(qrow + 16 + kb);
    qf = CONCAT16(lo, hi);
  }

  v8f out0 = {}, out1 = {};   // rows rbase+r, head-dim cols lm and lm+16

  for (int j = 0; j < 3; j++) {
    const _Float16* Kb = kws + (size_t)((j * BATCH + b) * NH + h) * SEQ * HD;
    const _Float16* Vb = vws + (size_t)((j * BATCH + b) * NH + h) * SEQ * HD;

    float m[8], l[8];
    #pragma unroll
    for (int r = 0; r < 8; r++) { m[r] = -3.0e38f; l[r] = 0.0f; }
    v8f acc0 = {}, acc1 = {};

    // prime the pipeline: async-copy chunk 0 into buffer 0
    async_copy_b128(ldsK[0], Kb + (size_t)crow * HD + csub);
    async_copy_b128(ldsV[0], Vb + (size_t)crow * SEQ + csub);

    for (int t0 = 0, it = 0; t0 < SEQ; t0 += 32, ++it) {
      const int buf = it & 1;
      const bool more = (t0 + 32) < SEQ;
      if (more) {  // overlap next chunk's DMA with this chunk's math
        async_copy_b128(ldsK[buf ^ 1], Kb + (size_t)(t0 + 32 + crow) * HD + csub);
        async_copy_b128(ldsV[buf ^ 1], Vb + (size_t)crow * SEQ + t0 + 32 + csub);
        asm volatile("s_wait_asynccnt 0x2" ::: "memory");  // current chunk landed
      } else {
        asm volatile("s_wait_asynccnt 0x0" ::: "memory");
      }
      __syncthreads();

      const _Float16 (*Kt)[40]  = Ktb[buf];
      const _Float16 (*VtT)[40] = VtTb[buf];

      // S = Q K^T over this 32-key chunk: keys 0-15 -> s0, 16-31 -> s1
      v8f s0 = {}, s1 = {};
      {
        const _Float16* kr0 = &Kt[lm][0];
        v8h lo = *(const v8h*)(kr0 + kb);
        v8h hi = *(const v8h*)(kr0 + 16 + kb);
        v16h bf = CONCAT16(lo, hi);
        s0 = __builtin_amdgcn_wmma_f32_16x16x32_f16(false, qf, false, bf, (short)0, s0, false, false);
        const _Float16* kr1 = &Kt[16 + lm][0];
        lo = *(const v8h*)(kr1 + kb);
        hi = *(const v8h*)(kr1 + 16 + kb);
        bf = CONCAT16(lo, hi);
        s1 = __builtin_amdgcn_wmma_f32_16x16x32_f16(false, qf, false, bf, (short)0, s1, false, false);
      }

      // online softmax update (row stats via 16-lane butterfly in each half)
      #pragma unroll
      for (int r = 0; r < 8; r++) {
        float mx = fmaxf(s0[r], s1[r]);
        mx = fmaxf(mx, __shfl_xor(mx, 1));
        mx = fmaxf(mx, __shfl_xor(mx, 2));
        mx = fmaxf(mx, __shfl_xor(mx, 4));
        mx = fmaxf(mx, __shfl_xor(mx, 8));
        float mn = fmaxf(m[r], mx);
        float cf = __expf(m[r] - mn);
        m[r] = mn;
        float p0 = __expf(s0[r] - mn);
        float p1 = __expf(s1[r] - mn);
        float ps = p0 + p1;
        ps += __shfl_xor(ps, 1);
        ps += __shfl_xor(ps, 2);
        ps += __shfl_xor(ps, 4);
        ps += __shfl_xor(ps, 8);
        l[r] = l[r] * cf + ps;
        acc0[r] *= cf;
        acc1[r] *= cf;
        union { _Float16 hh[2]; unsigned u; } pk;   // keys (lm, lm+16) packed
        pk.hh[0] = (_Float16)p0;
        pk.hh[1] = (_Float16)p1;
        Pt[wave][rbase + r][lm] = pk.u;
      }
      // Pt is wave-private; same-wave LDS ops are in-order -> counter wait only
      asm volatile("s_wait_dscnt 0x0" ::: "memory");

      // acc += P (16x32 keys, permuted packing) * V (keys x 32 dims)
      {
        const _Float16* prow = (const _Float16*)&Pt[wave][lm][0];
        v8h plo = *(const v8h*)(prow + kb2);
        v8h phi = *(const v8h*)(prow + kb2 + 8);
        v16h pf = CONCAT16(plo, phi);

        const _Float16* vr0 = &VtT[lm][0];          // dims 0-15
        v8h v0 = *(const v8h*)(vr0 + kb);
        v8h v1 = *(const v8h*)(vr0 + 16 + kb);
        v16h bf = ILV16(v0, v1);                    // matches P's key permutation
        acc0 = __builtin_amdgcn_wmma_f32_16x16x32_f16(false, pf, false, bf, (short)0, acc0, false, false);

        const _Float16* vr1 = &VtT[16 + lm][0];     // dims 16-31
        v0 = *(const v8h*)(vr1 + kb);
        v1 = *(const v8h*)(vr1 + 16 + kb);
        bf = ILV16(v0, v1);
        acc1 = __builtin_amdgcn_wmma_f32_16x16x32_f16(false, pf, false, bf, (short)0, acc1, false, false);
      }
      __syncthreads();  // all waves done with this buffer before DMA reuses it
    }

    // out_i += normalized attention for pair (i, j)
    #pragma unroll
    for (int r = 0; r < 8; r++) {
      float inv = 1.0f / l[r];
      out0[r] += acc0[r] * inv;
      out1[r] += acc1[r] * inv;
    }
  }

  // write O in [asset][B][S][DM] f32 for the output projection
  #pragma unroll
  for (int r = 0; r < 8; r++) {
    int srow = q0 + wave * 16 + rbase + r;
    size_t base = ((size_t)(i * BATCH + b) * SEQ + srow) * DM + h * HD;
    ows[base + lm]      = out0[r];
    ows[base + 16 + lm] = out1[r];
  }
}

// ---------------------------------------------------------------------------
// Phase 3: out = O @ Wo + bo.  M = 3*B*S = 12288, N = K = 256, f32 -> d_out
// (tuple (out0,out1,out2) is exactly the contiguous [3][B][S][DM] flattening).
// ---------------------------------------------------------------------------
__global__ __launch_bounds__(128) void out_proj_kernel(
    const float* __restrict__ O, const float* __restrict__ Wo,
    const float* __restrict__ bo, float* __restrict__ out)
{
  __shared__ __align__(16) _Float16 As[128][40];
  __shared__ __align__(16) _Float16 BsT[64][40];

  const int m0 = blockIdx.x * 128;
  const int n0 = blockIdx.y * 64;
  const int t = threadIdx.x;
  const int wave = t >> 5, lane = t & 31;
  const int lm = lane & 15, rbase = (lane >> 4) * 8;

  v8f c[2][4] = {};
  gemm128x64(O, Wo, m0, n0, DM, DM, DM, t, c, As, BsT);

  #pragma unroll
  for (int s = 0; s < 2; s++) {
    #pragma unroll
    for (int ns = 0; ns < 4; ns++) {
      #pragma unroll
      for (int r = 0; r < 8; r++) {
        int row = m0 + wave * 32 + s * 16 + rbase + r;
        int col = n0 + ns * 16 + lm;
        out[(size_t)row * DM + col] = c[s][ns][r] + bo[col];
      }
    }
  }
}

// ---------------------------------------------------------------------------
extern "C" void kernel_launch(void* const* d_in, const int* in_sizes, int n_in,
                              void* d_out, int out_size, void* d_ws, size_t ws_size,
                              hipStream_t stream) {
  const float* x0 = (const float*)d_in[0];
  const float* x1 = (const float*)d_in[1];
  const float* x2 = (const float*)d_in[2];
  const float* Wq = (const float*)d_in[3];
  const float* bq = (const float*)d_in[4];
  const float* Wk = (const float*)d_in[5];
  const float* bk = (const float*)d_in[6];
  const float* Wv = (const float*)d_in[7];
  const float* bv = (const float*)d_in[8];
  const float* Wo = (const float*)d_in[9];
  const float* bo = (const float*)d_in[10];

  // workspace: Q/K f16 [3][B][H][S][HD], V f16 transposed [3][B][H][HD][S],
  // then O f32 [3][B][S][DM]
  const size_t QKV_ELEMS = (size_t)3 * BATCH * NH * SEQ * HD;  // 3,145,728
  _Float16* qws = (_Float16*)d_ws;
  _Float16* kws = qws + QKV_ELEMS;
  _Float16* vws = kws + QKV_ELEMS;
  float*    ows = (float*)(vws + QKV_ELEMS);

  // Phase 1: 9 projection GEMMs (asset x {q,k,v}), M=4096, N=256
  dim3 g1(BATCH * SEQ / 128, DM / 64, 9);
  qkv_proj_kernel<<<g1, 128, 0, stream>>>(x0, x1, x2, Wq, bq, Wk, bk, Wv, bv,
                                          qws, kws, vws);

  // Phase 2: fused cross flash-attention (async double-buffered K/V)
  dim3 g2(SEQ / 64, BATCH * NH, 3);
  cross_attn_kernel<<<g2, 128, 0, stream>>>(qws, kws, vws, ows);

  // Phase 3: output projection, M = 3*B*S = 12288
  dim3 g3(3 * BATCH * SEQ / 128, DM / 64, 1);
  out_proj_kernel<<<g3, 128, 0, stream>>>(ows, Wo, bo, (float*)d_out);
}
